// DGLTemporalAttention_5866925326564
// MI455X (gfx1250) — matrix operand order repaired and checked by reading, not compiled, not run on hardware
//
#include <hip/hip_runtime.h>

#define D_NODES 50000
#define E_EDGES 800000
#define EPS_LN 1e-5f

typedef __bf16 bf16_t;
typedef bf16_t v16bf __attribute__((ext_vector_type(16)));
typedef bf16_t v8bf  __attribute__((ext_vector_type(8)));
typedef float  v8f   __attribute__((ext_vector_type(8)));

// ---- order-preserving f32 <-> u32 encoding (for atomicMax on signed floats) ----
__device__ __forceinline__ unsigned enc_f32(float x) {
    unsigned b = __float_as_uint(x);
    return (b & 0x80000000u) ? ~b : (b | 0x80000000u);
}
__device__ __forceinline__ float dec_f32(unsigned k) {
    return (k & 0x80000000u) ? __uint_as_float(k & 0x7FFFFFFFu) : __uint_as_float(~k);
}
#define ENC_NEG_INF 0x007FFFFFu  // enc_f32(-inf)

// ---- A-fragment loader: contiguous K order (matches B) ----
// lanes 0-15  : K = {k0..k0+15}
// lanes 16-31 : K = {k0+16..k0+31}
__device__ __forceinline__ v16bf load_afrag(const bf16_t* chunk, int lane) {
    int sub = (lane & 16) ? 16 : 0;
    v8bf lo = *(const v8bf*)(chunk + sub);
    v8bf hi = *(const v8bf*)(chunk + 8 + sub);
    return __builtin_shufflevector(lo, hi, 0,1,2,3,4,5,6,7,8,9,10,11,12,13,14,15);
}

// ================= init =================
__global__ void k_init(float* denom, float* agg, unsigned* segmax) {
    long i = (long)blockIdx.x * 256 + threadIdx.x;
    if (i < (long)D_NODES * 8) { denom[i] = 0.f; segmax[i] = ENC_NEG_INF; }
    if (i < (long)D_NODES * 128) agg[i] = 0.f;
}

__global__ void k_cvt(const float* __restrict__ src, bf16_t* __restrict__ dst, int n) {
    int i = blockIdx.x * 256 + threadIdx.x;
    if (i < n) dst[i] = (bf16_t)src[i];
}

// wq is [128 x 256]; keep only node part (first 128 K-cols) as bf16 [128 x 128]
__global__ void k_cvt_wq(const float* __restrict__ wq, bf16_t* __restrict__ dst) {
    int i = blockIdx.x * 256 + threadIdx.x;
    if (i < 128 * 128) dst[i] = (bf16_t)wq[(i >> 7) * 256 + (i & 127)];
}

// constq[o] = bq[o] + sum_t cos(tb[t]) * wq[o, 128+t]
__global__ void k_constq(const float* __restrict__ wq, const float* __restrict__ bq,
                         const float* __restrict__ tb, float* __restrict__ constq) {
    int o = threadIdx.x;
    float s = bq[o];
    for (int t = 0; t < 128; ++t) s += __cosf(tb[t]) * wq[o * 256 + 128 + t];
    constq[o] = s;
}

// ================= Q projection: q_nodes[D,128] = h[:D] @ wq[:, :128]^T + constq =====
__global__ __launch_bounds__(32) void k_q(const float* __restrict__ h,
                                          const bf16_t* __restrict__ wq_bf,
                                          const float* __restrict__ constq,
                                          float* __restrict__ q_nodes) {
    const int lane = threadIdx.x;
    const long d0 = (long)blockIdx.x * 16;
    __shared__ __align__(32) bf16_t atile[16 * 136];  // 128 + 8 pad

    for (int row = 0; row < 16; ++row) {
        const float* hr = h + (d0 + row) * 128;
        bf16_t* dr = atile + row * 136;
        for (int col = lane; col < 128; col += 32) dr[col] = (bf16_t)hr[col];
    }
    __syncthreads();

    const int arow = lane & 15;
    v16bf afrag[4];
    #pragma unroll
    for (int kc = 0; kc < 4; ++kc)
        afrag[kc] = load_afrag(atile + arow * 136 + kc * 32, lane);

    const int bn = lane & 15;
    const int bkof = (lane & 16) ? 16 : 0;
    const int mrow = (lane & 16) ? 8 : 0;

    for (int nt = 0; nt < 8; ++nt) {
        v8f acc = {};
        const bf16_t* brow = wq_bf + (long)(nt * 16 + bn) * 128 + bkof;
        #pragma unroll
        for (int kc = 0; kc < 4; ++kc) {
            v16bf b = *(const v16bf*)(brow + kc * 32);
            acc = __builtin_amdgcn_wmma_f32_16x16x32_bf16(false, afrag[kc], false, b,
                                                          (short)0, acc, false, false);
        }
        int coln = nt * 16 + bn;
        float cq = constq[coln];
        #pragma unroll
        for (int r = 0; r < 8; ++r)
            q_nodes[(d0 + mrow + r) * 128 + coln] = acc[r] + cq;
    }
}

// ================= K/V projections + attention scores (per 16-edge tile) =============
__global__ __launch_bounds__(32) void k_kv(const float* __restrict__ h,
                                           const float* __restrict__ f,
                                           const float* __restrict__ dt,
                                           const int* __restrict__ dst_idx,
                                           const float* __restrict__ tw,
                                           const float* __restrict__ tb,
                                           const float* __restrict__ bk,
                                           const float* __restrict__ bv,
                                           const bf16_t* __restrict__ wk_bf,
                                           const bf16_t* __restrict__ wv_bf,
                                           const float* __restrict__ q_nodes,
                                           float* __restrict__ Vout,
                                           float* __restrict__ score,
                                           unsigned* __restrict__ segmax) {
    const int lane = threadIdx.x;
    const long e0 = (long)blockIdx.x * 16;
    __shared__ __align__(32) unsigned char smem[16 * 392 * 2];  // 12544 B
    bf16_t* akv = (bf16_t*)smem;   // [16][392] bf16 A tile
    float*  kbuf = (float*)smem;   // [16][132] f32 K tile (aliased, used later)

    // Build A tile: [h(D+e) | f(e) | cos(dt*tw + tb)] -> bf16
    for (int row = 0; row < 16; ++row) {
        long e = e0 + row;
        float trow = dt[e];
        const float* hr = h + (long)(D_NODES + e) * 128;
        const float* fr = f + e * 128;
        bf16_t* dr = akv + row * 392;
        for (int col = lane; col < 384; col += 32) {
            float v;
            if (col < 128)       v = hr[col];
            else if (col < 256)  v = fr[col - 128];
            else                 v = __cosf(trow * tw[col - 256] + tb[col - 256]);
            dr[col] = (bf16_t)v;
        }
    }
    __syncthreads();

    const int arow = lane & 15;
    v16bf afrag[12];
    #pragma unroll
    for (int kc = 0; kc < 12; ++kc)
        afrag[kc] = load_afrag(akv + arow * 392 + kc * 32, lane);
    __syncthreads();

    const int bn = lane & 15;
    const int bkof = (lane & 16) ? 16 : 0;
    const int mrow = (lane & 16) ? 8 : 0;

    // K = A @ wk^T + bk -> LDS (overwrites A tile; A frags already in VGPRs)
    for (int nt = 0; nt < 8; ++nt) {
        v8f acc = {};
        const bf16_t* brow = wk_bf + (long)(nt * 16 + bn) * 384 + bkof;
        #pragma unroll
        for (int kc = 0; kc < 12; ++kc) {
            v16bf b = *(const v16bf*)(brow + kc * 32);
            acc = __builtin_amdgcn_wmma_f32_16x16x32_bf16(false, afrag[kc], false, b,
                                                          (short)0, acc, false, false);
        }
        int coln = nt * 16 + bn;
        float bias = bk[coln];
        #pragma unroll
        for (int r = 0; r < 8; ++r)
            kbuf[(mrow + r) * 132 + coln] = acc[r] + bias;
    }
    __syncthreads();

    // scores: 16 edges x 8 heads; Q gathered from L2-resident q_nodes
    #pragma unroll
    for (int i = 0; i < 4; ++i) {
        int p = lane + 32 * i;     // 0..127
        int m = p >> 3;
        int hh = p & 7;
        long e = e0 + m;
        int dsti = dst_idx[e];
        const float* qrow = q_nodes + (long)dsti * 128 + hh * 16;
        const float* krow = kbuf + m * 132 + hh * 16;
        float d = 0.f;
        #pragma unroll
        for (int j = 0; j < 16; ++j) d += qrow[j] * krow[j];
        float s = d > 0.f ? d : 0.2f * d;       // leaky_relu(0.2)
        score[e * 8 + hh] = s;
        atomicMax(segmax + (long)dsti * 8 + hh, enc_f32(s));
    }

    // V = A @ wv^T + bv -> workspace
    for (int nt = 0; nt < 8; ++nt) {
        v8f acc = {};
        const bf16_t* brow = wv_bf + (long)(nt * 16 + bn) * 384 + bkof;
        #pragma unroll
        for (int kc = 0; kc < 12; ++kc) {
            v16bf b = *(const v16bf*)(brow + kc * 32);
            acc = __builtin_amdgcn_wmma_f32_16x16x32_bf16(false, afrag[kc], false, b,
                                                          (short)0, acc, false, false);
        }
        int coln = nt * 16 + bn;
        float bias = bv[coln];
        #pragma unroll
        for (int r = 0; r < 8; ++r)
            Vout[(e0 + mrow + r) * 128 + coln] = acc[r] + bias;
    }
}

// ================= softmax numerator + segment sums (att = ex/denom folded later) ====
__global__ void k_sm(const int* __restrict__ dst_idx, const float* __restrict__ score,
                     const unsigned* __restrict__ segmax, const float* __restrict__ Vws,
                     float* __restrict__ denom, float* __restrict__ agg) {
    long gid = (long)blockIdx.x * 256 + threadIdx.x;   // e*128 + col
    if (gid >= (long)E_EDGES * 128) return;
    long e = gid >> 7;
    int col = (int)(gid & 127);
    int hh = col >> 4;
    int dsti = dst_idx[e];
    float m = dec_f32(segmax[(long)dsti * 8 + hh]);
    float ex = __expf(score[e * 8 + hh] - m);
    if ((col & 15) == 0) atomicAdd(denom + (long)dsti * 8 + hh, ex);
    atomicAdd(agg + (long)dsti * 128 + col, ex * Vws[gid]);
}

// ================= output projection + ReLU + LayerNorm (per 16-node tile) ===========
__global__ __launch_bounds__(32) void k_out(const float* __restrict__ h,
                                            const float* __restrict__ agg,
                                            const float* __restrict__ denom,
                                            const bf16_t* __restrict__ wout_bf,
                                            const float* __restrict__ bout,
                                            const float* __restrict__ gamma,
                                            const float* __restrict__ beta,
                                            float* __restrict__ out) {
    const int lane = threadIdx.x;
    const long d0 = (long)blockIdx.x * 16;
    __shared__ __align__(32) unsigned char smem[16 * 264 * 2];  // 8448 B
    bf16_t* ain = (bf16_t*)smem;  // [16][264]
    float*  rst = (float*)smem;   // [16][132] (aliased)

    for (int row = 0; row < 16; ++row) {
        long d = d0 + row;
        bf16_t* dr = ain + row * 264;
        for (int col = lane; col < 256; col += 32) {
            float v;
            if (col < 128) {
                float dn = denom[d * 8 + (col >> 4)];
                v = (dn != 0.f) ? agg[d * 128 + col] / dn : 0.f;
            } else {
                v = h[d * 128 + (col - 128)];
            }
            dr[col] = (bf16_t)v;
        }
    }
    __syncthreads();

    const int arow = lane & 15;
    v16bf afrag[8];
    #pragma unroll
    for (int kc = 0; kc < 8; ++kc)
        afrag[kc] = load_afrag(ain + arow * 264 + kc * 32, lane);
    __syncthreads();

    const int bn = lane & 15;
    const int bkof = (lane & 16) ? 16 : 0;
    const int mrow = (lane & 16) ? 8 : 0;

    for (int nt = 0; nt < 8; ++nt) {
        v8f acc = {};
        const bf16_t* brow = wout_bf + (long)(nt * 16 + bn) * 256 + bkof;
        #pragma unroll
        for (int kc = 0; kc < 8; ++kc) {
            v16bf b = *(const v16bf*)(brow + kc * 32);
            acc = __builtin_amdgcn_wmma_f32_16x16x32_bf16(false, afrag[kc], false, b,
                                                          (short)0, acc, false, false);
        }
        int coln = nt * 16 + bn;
        float bias = bout[coln];
        #pragma unroll
        for (int r = 0; r < 8; ++r)
            rst[(mrow + r) * 132 + coln] = fmaxf(acc[r] + bias, 0.f);
    }
    __syncthreads();

    // LayerNorm: 2 lanes per row, 64 cols each
    int row = lane >> 1, half = lane & 1;
    const float* rr = rst + row * 132 + half * 64;
    float s = 0.f, s2 = 0.f;
    #pragma unroll 8
    for (int j = 0; j < 64; ++j) { float x = rr[j]; s += x; s2 += x * x; }
    s  += __shfl_xor(s, 1);
    s2 += __shfl_xor(s2, 1);
    float mu  = s * (1.f / 128.f);
    float var = s2 * (1.f / 128.f) - mu * mu;
    float rs  = rsqrtf(var + EPS_LN);
    #pragma unroll 8
    for (int j = 0; j < 64; ++j) {
        int c = half * 64 + j;
        out[(d0 + row) * 128 + c] = (rr[j] - mu) * rs * gamma[c] + beta[c];
    }
}

// ================= host launcher =================
extern "C" void kernel_launch(void* const* d_in, const int* in_sizes, int n_in,
                              void* d_out, int out_size, void* d_ws, size_t ws_size,
                              hipStream_t stream) {
    const float* h    = (const float*)d_in[0];
    const float* f    = (const float*)d_in[1];
    const float* dt   = (const float*)d_in[2];
    const int*   dsti = (const int*)d_in[3];
    const float* tw   = (const float*)d_in[4];
    const float* tb   = (const float*)d_in[5];
    const float* wq   = (const float*)d_in[6];
    const float* bq   = (const float*)d_in[7];
    const float* wk   = (const float*)d_in[8];
    const float* bk   = (const float*)d_in[9];
    const float* wv   = (const float*)d_in[10];
    const float* bv   = (const float*)d_in[11];
    const float* wout = (const float*)d_in[12];
    const float* bout = (const float*)d_in[13];
    const float* gam  = (const float*)d_in[14];
    const float* bet  = (const float*)d_in[15];
    float* out = (float*)d_out;

    unsigned char* ws = (unsigned char*)d_ws;
    size_t off = 0;
    auto take = [&](size_t bytes) { size_t o = off; off += (bytes + 255) & ~(size_t)255; return o; };
    float*    q_nodes = (float*)   (ws + take((size_t)D_NODES * 128 * 4));
    float*    Vws     = (float*)   (ws + take((size_t)E_EDGES * 128 * 4));
    float*    score   = (float*)   (ws + take((size_t)E_EDGES * 8 * 4));
    unsigned* segmax  = (unsigned*)(ws + take((size_t)D_NODES * 8 * 4));
    float*    denom   = (float*)   (ws + take((size_t)D_NODES * 8 * 4));
    float*    agg     = (float*)   (ws + take((size_t)D_NODES * 128 * 4));
    bf16_t*   wq_bf   = (bf16_t*)  (ws + take(128 * 128 * 2));
    bf16_t*   wk_bf   = (bf16_t*)  (ws + take(128 * 384 * 2));
    bf16_t*   wv_bf   = (bf16_t*)  (ws + take(128 * 384 * 2));
    bf16_t*   wout_bf = (bf16_t*)  (ws + take(128 * 256 * 2));
    float*    constq  = (float*)   (ws + take(128 * 4));

    // 1) init accumulators / segment-max
    k_init<<<(D_NODES * 128 + 255) / 256, 256, 0, stream>>>(denom, agg, segmax);
    // 2) weight conversion to bf16
    k_cvt<<<(128 * 384 + 255) / 256, 256, 0, stream>>>(wk, wk_bf, 128 * 384);
    k_cvt<<<(128 * 384 + 255) / 256, 256, 0, stream>>>(wv, wv_bf, 128 * 384);
    k_cvt<<<(128 * 256 + 255) / 256, 256, 0, stream>>>(wout, wout_bf, 128 * 256);
    k_cvt_wq<<<(128 * 128 + 255) / 256, 256, 0, stream>>>(wq, wq_bf);
    k_constq<<<1, 128, 0, stream>>>(wq, bq, tb, constq);
    // 3) Q projection (WMMA bf16)
    k_q<<<D_NODES / 16, 32, 0, stream>>>(h, wq_bf, constq, q_nodes);
    // 4) K/V projections + scores + segment max (WMMA bf16)
    k_kv<<<E_EDGES / 16, 32, 0, stream>>>(h, f, dt, dsti, tw, tb, bk, bv,
                                          wk_bf, wv_bf, q_nodes, Vws, score, segmax);
    // 5) softmax numerators + segment sums
    k_sm<<<(int)(((long)E_EDGES * 128 + 255) / 256), 256, 0, stream>>>(dsti, score, segmax,
                                                                      Vws, denom, agg);
    // 6) output projection + ReLU + LayerNorm (WMMA bf16)
    k_out<<<D_NODES / 16, 32, 0, stream>>>(h, agg, denom, wout_bf, bout, gam, bet, out);
}
